// selfAttention_11166914970432
// MI455X (gfx1250) — matrix operand order repaired and checked by reading, not compiled, not run on hardware
//
#include <hip/hip_runtime.h>

typedef __bf16 bf16_t;
typedef __attribute__((ext_vector_type(16))) __bf16 v16bf;
typedef __attribute__((ext_vector_type(8)))  __bf16 v8bf;
typedef __attribute__((ext_vector_type(4)))  __bf16 v4bf;
typedef __attribute__((ext_vector_type(8)))  float  v8f;

#define B_  2
#define S_  2048
#define D_  1024
#define H_  16
#define DH_ 64

#define WMMA_BF16(a, b, c) \
  __builtin_amdgcn_wmma_f32_16x16x32_bf16(false, (a), false, (b), (short)0, (c), false, false)

// ---------- WMMA fragment loaders (wave32, CDNA5 layouts) ----------

// A-fragment: 16(M) x 32(K) bf16 tile, row-major, leading dim `ld`.
// lane&15 = row M; lanes 0-15 hold K=0..15; lanes 16-31 hold K=16..31.
__device__ __forceinline__ v16bf load_a16x32(const bf16_t* __restrict__ tile, int ld) {
  const int lane = threadIdx.x & 31;
  const bf16_t* p = tile + (lane & 15) * ld + ((lane >> 4) << 4);
  v8bf lo = *(const v8bf*)(p);
  v8bf hi = *(const v8bf*)(p + 8);
  v16bf a;
#pragma unroll
  for (int i = 0; i < 8; ++i) { a[i] = lo[i]; a[i + 8] = hi[i]; }
  return a;
}

// B-fragment: 32(K) x 16(N) bf16 from transposed storage Bt[N][K]; one 32B load/lane.
__device__ __forceinline__ v16bf load_b32x16_t(const bf16_t* __restrict__ btile, int ldb) {
  const int lane = threadIdx.x & 31;
  return *(const v16bf*)(btile + (lane & 15) * ldb + ((lane >> 4) << 4));
}

// ---------- pre-pass: fp32 -> bf16 convert / transpose ----------

__global__ void to_bf16_kernel(const float* __restrict__ in, bf16_t* __restrict__ out, int n4) {
  int i = blockIdx.x * blockDim.x + threadIdx.x;
  if (i >= n4) return;
  float4 v = ((const float4*)in)[i];
  v4bf o = { (bf16_t)v.x, (bf16_t)v.y, (bf16_t)v.z, (bf16_t)v.w };
  ((v4bf*)out)[i] = o;
}

// in[rows][cols] fp32 -> out[cols][rows] bf16 (LDS-tiled, conflict-free)
__global__ void transpose_to_bf16(const float* __restrict__ in, bf16_t* __restrict__ out,
                                  int rows, int cols) {
  __shared__ float tile[32][33];
  const int c0 = blockIdx.x * 32, r0 = blockIdx.y * 32;
#pragma unroll
  for (int i = threadIdx.y; i < 32; i += 8)
    tile[i][threadIdx.x] = in[(size_t)(r0 + i) * cols + (c0 + threadIdx.x)];
  __syncthreads();
#pragma unroll
  for (int i = threadIdx.y; i < 32; i += 8)
    out[(size_t)(c0 + i) * rows + (r0 + threadIdx.x)] = (bf16_t)tile[threadIdx.x][i];
}

// ---------- QKV scatter-store of one 16x16 fp32 tile ----------

__device__ __forceinline__ void store_qkv_tile(const v8f& acc, int m0, int n,
                                               bf16_t* __restrict__ Qb,
                                               bf16_t* __restrict__ Kb,
                                               bf16_t* __restrict__ Vt) {
  const int lane = threadIdx.x & 31;
  const int hv = lane >> 4;
  const int sel = n >> 10;             // 0=Q 1=K 2=V
  const int cc  = n & 1023;
  const int h   = cc >> 6, dh = cc & 63;
#pragma unroll
  for (int r = 0; r < 8; ++r) {
    const int m  = m0 + (hv << 3) + r;
    const int bi = m >> 11, s = m & 2047;
    const bf16_t v = (bf16_t)acc[r];
    if (sel == 0)      Qb[(((size_t)(bi * H_ + h)) * S_ + s) * DH_ + dh] = v;
    else if (sel == 1) Kb[(((size_t)(bi * H_ + h)) * S_ + s) * DH_ + dh] = v;
    else               Vt[(((size_t)(bi * H_ + h)) * DH_ + dh) * S_ + s] = v;
  }
}

// ---------- kernel 1: QKV projection, 16x64 tile per wave, double-buffered ----------

__global__ void __launch_bounds__(128)
qkv_gemm_kernel(const bf16_t* __restrict__ xb, const bf16_t* __restrict__ WinT,
                bf16_t* __restrict__ Qb, bf16_t* __restrict__ Kb, bf16_t* __restrict__ Vt) {
  const int wid = blockIdx.x * 4 + (threadIdx.x >> 5);
  const int nt = wid % 48;             // 3D/64 = 48 n-supertiles
  const int mt = wid / 48;             // B*S/16 = 256 m-tiles
  const int m0 = mt << 4, n0 = nt << 6;

  const bf16_t* a_base = xb   + (size_t)m0 * D_;
  const bf16_t* b_base = WinT + (size_t)n0 * D_;

  v8f acc0 = {}, acc1 = {}, acc2 = {}, acc3 = {};

  // prologue: stage k=0 fragments
  v16bf av = load_a16x32(a_base, D_);
  v16bf b0 = load_b32x16_t(b_base, D_);
  v16bf b1 = load_b32x16_t(b_base + (size_t)16 * D_, D_);
  v16bf b2 = load_b32x16_t(b_base + (size_t)32 * D_, D_);
  v16bf b3 = load_b32x16_t(b_base + (size_t)48 * D_, D_);

  for (int k = 32; k < D_; k += 32) {
    // issue next k-step loads BEFORE consuming current fragments
    v16bf avn = load_a16x32(a_base + k, D_);
    v16bf b0n = load_b32x16_t(b_base + k, D_);
    v16bf b1n = load_b32x16_t(b_base + (size_t)16 * D_ + k, D_);
    v16bf b2n = load_b32x16_t(b_base + (size_t)32 * D_ + k, D_);
    v16bf b3n = load_b32x16_t(b_base + (size_t)48 * D_ + k, D_);
    acc0 = WMMA_BF16(av, b0, acc0);
    acc1 = WMMA_BF16(av, b1, acc1);
    acc2 = WMMA_BF16(av, b2, acc2);
    acc3 = WMMA_BF16(av, b3, acc3);
    av = avn; b0 = b0n; b1 = b1n; b2 = b2n; b3 = b3n;
  }
  acc0 = WMMA_BF16(av, b0, acc0);
  acc1 = WMMA_BF16(av, b1, acc1);
  acc2 = WMMA_BF16(av, b2, acc2);
  acc3 = WMMA_BF16(av, b3, acc3);

  const int colv = threadIdx.x & 15;
  store_qkv_tile(acc0, m0, n0 +  0 + colv, Qb, Kb, Vt);
  store_qkv_tile(acc1, m0, n0 + 16 + colv, Qb, Kb, Vt);
  store_qkv_tile(acc2, m0, n0 + 32 + colv, Qb, Kb, Vt);
  store_qkv_tile(acc3, m0, n0 + 48 + colv, Qb, Kb, Vt);
}

// ---------- kernel 2: causal flash attention (scores computed transposed) ----------

__global__ void __launch_bounds__(128)
attn_kernel(const bf16_t* __restrict__ Qb, const bf16_t* __restrict__ Kb,
            const bf16_t* __restrict__ Vt, bf16_t* __restrict__ O) {
  const int wid = blockIdx.x * 4 + (threadIdx.x >> 5);
  const int qt = wid & 127;            // S/16 q-tiles
  const int bh = wid >> 7;             // b*H + h
  const int qm = qt << 4;
  const int lane = threadIdx.x & 31;
  const int colv = lane & 15, hv = lane >> 4;

  const bf16_t* Qp = Qb + (size_t)bh * S_ * DH_;
  const bf16_t* Kp = Kb + (size_t)bh * S_ * DH_;
  const bf16_t* Vp = Vt + (size_t)bh * DH_ * S_;

  // Q^T B-fragments (col = q, K dim = dh), held across the whole key loop
  const bf16_t* qptr = Qp + (size_t)(qm + colv) * DH_ + (hv << 4);
  const v16bf bq0 = *(const v16bf*)(qptr);        // dh 0..31
  const v16bf bq1 = *(const v16bf*)(qptr + 32);   // dh 32..63

  float mi = -__builtin_inff();
  float li = 0.f;
  v8f acc0 = {}, acc1 = {}, acc2 = {}, acc3 = {};

  const int   jlast = qm & ~31;        // last (diagonal) 32-key slab
  const float scale = 0.125f;          // 1/sqrt(64)
  const int   q     = qm + colv;

  for (int jb = 0; jb <= jlast; jb += 32) {
    // Prefetch the NEXT slab of K/V into cache (hides L2 latency of next iter)
    if (jb + 32 <= jlast) {
      const bf16_t* nk = Kp + (size_t)(jb + 32) * DH_;
      __builtin_prefetch(nk + (size_t)lane * 64, 0, 3);          // 32 x 128B = K slab
      const bf16_t* nv = Vp + (jb + 32);
      __builtin_prefetch(nv + (size_t)(lane * 2 + 0) * S_, 0, 3);
      __builtin_prefetch(nv + (size_t)(lane * 2 + 1) * S_, 0, 3);
    }

    // Issue ALL K and V fragment loads up front (distinct regs -> one clause,
    // V loads remain in flight across the whole softmax VALU chain).
    v16bf ak00 = load_a16x32(Kp + (size_t)(jb +  0) * DH_ +  0, DH_);
    v16bf ak01 = load_a16x32(Kp + (size_t)(jb +  0) * DH_ + 32, DH_);
    v16bf ak10 = load_a16x32(Kp + (size_t)(jb + 16) * DH_ +  0, DH_);
    v16bf ak11 = load_a16x32(Kp + (size_t)(jb + 16) * DH_ + 32, DH_);
    v16bf av0  = load_a16x32(Vp + (size_t)( 0) * S_ + jb, S_);
    v16bf av1  = load_a16x32(Vp + (size_t)(16) * S_ + jb, S_);
    v16bf av2  = load_a16x32(Vp + (size_t)(32) * S_ + jb, S_);
    v16bf av3  = load_a16x32(Vp + (size_t)(48) * S_ + jb, S_);

    // scores^T = K · Q^T : two 16(key) x 16(q) fp32 tiles
    v8f c0 = {}, c1 = {};
    c0 = WMMA_BF16(ak00, bq0, c0);
    c0 = WMMA_BF16(ak01, bq1, c0);
    c1 = WMMA_BF16(ak10, bq0, c1);
    c1 = WMMA_BF16(ak11, bq1, c1);

#pragma unroll
    for (int r = 0; r < 8; ++r) {
      c0[r] *= scale; c1[r] *= scale;
      if (jb == jlast) {               // causal mask only on diagonal slab
        const int key0 = jb + (hv << 3) + r;
        if (key0 > q)      c0[r] = -__builtin_inff();
        if (key0 + 16 > q) c1[r] = -__builtin_inff();
      }
    }

    // online softmax: stats per q-column live in lane pair (l, l^16)
    float tmax = -__builtin_inff();
#pragma unroll
    for (int r = 0; r < 8; ++r) tmax = fmaxf(tmax, fmaxf(c0[r], c1[r]));
    tmax = fmaxf(tmax, __shfl_xor(tmax, 16, 32));
    const float mnew = fmaxf(mi, tmax);
    const float corr = __expf(mi - mnew);
    float p0[8], p1[8], tsum = 0.f;
#pragma unroll
    for (int r = 0; r < 8; ++r) {
      p0[r] = __expf(c0[r] - mnew);
      p1[r] = __expf(c1[r] - mnew);
      tsum += p0[r] + p1[r];
    }
    tsum += __shfl_xor(tsum, 16, 32);
    li = li * corr + tsum;
    mi = mnew;
#pragma unroll
    for (int r = 0; r < 8; ++r) { acc0[r] *= corr; acc1[r] *= corr; acc2[r] *= corr; acc3[r] *= corr; }

    // Repack P^T (32 keys x 16 q) C-layout -> B-fragment: one lane-pair exchange
    v16bf pb;
#pragma unroll
    for (int r = 0; r < 8; ++r) {
      const float o0 = __shfl_xor(p0[r], 16, 32);
      const float o1 = __shfl_xor(p1[r], 16, 32);
      if (hv == 0) { pb[r] = (bf16_t)p0[r]; pb[r + 8] = (bf16_t)o0;    }
      else         { pb[r] = (bf16_t)o1;    pb[r + 8] = (bf16_t)p1[r]; }
    }

    // out^T += V^T · P^T   (4 dh-tiles of 16); V fragments already resident
    acc0 = WMMA_BF16(av0, pb, acc0);
    acc1 = WMMA_BF16(av1, pb, acc1);
    acc2 = WMMA_BF16(av2, pb, acc2);
    acc3 = WMMA_BF16(av3, pb, acc3);
  }

  const float inv = 1.f / li;
  const int bi = bh >> 4, h = bh & 15;
  bf16_t* op = O + ((size_t)(bi * S_ + qm + colv)) * D_ + h * DH_;
#pragma unroll
  for (int r = 0; r < 8; ++r) {
    const int d = (hv << 3) + r;
    op[d +  0] = (bf16_t)(acc0[r] * inv);
    op[d + 16] = (bf16_t)(acc1[r] * inv);
    op[d + 32] = (bf16_t)(acc2[r] * inv);
    op[d + 48] = (bf16_t)(acc3[r] * inv);
  }
}

// ---------- kernel 3: output projection, 16x64 tile per wave, double-buffered ----------

__global__ void __launch_bounds__(128)
out_proj_kernel(const bf16_t* __restrict__ Ob, const bf16_t* __restrict__ WoutT,
                float* __restrict__ out) {
  const int wid = blockIdx.x * 4 + (threadIdx.x >> 5);
  const int nt = wid & 15;             // D/64 = 16 n-supertiles
  const int mt = wid >> 4;             // B*S/16 = 256 m-tiles
  const int m0 = mt << 4, n0 = nt << 6;

  const bf16_t* a_base = Ob    + (size_t)m0 * D_;
  const bf16_t* b_base = WoutT + (size_t)n0 * D_;

  v8f acc0 = {}, acc1 = {}, acc2 = {}, acc3 = {};

  v16bf av = load_a16x32(a_base, D_);
  v16bf b0 = load_b32x16_t(b_base, D_);
  v16bf b1 = load_b32x16_t(b_base + (size_t)16 * D_, D_);
  v16bf b2 = load_b32x16_t(b_base + (size_t)32 * D_, D_);
  v16bf b3 = load_b32x16_t(b_base + (size_t)48 * D_, D_);

  for (int k = 32; k < D_; k += 32) {
    v16bf avn = load_a16x32(a_base + k, D_);
    v16bf b0n = load_b32x16_t(b_base + k, D_);
    v16bf b1n = load_b32x16_t(b_base + (size_t)16 * D_ + k, D_);
    v16bf b2n = load_b32x16_t(b_base + (size_t)32 * D_ + k, D_);
    v16bf b3n = load_b32x16_t(b_base + (size_t)48 * D_ + k, D_);
    acc0 = WMMA_BF16(av, b0, acc0);
    acc1 = WMMA_BF16(av, b1, acc1);
    acc2 = WMMA_BF16(av, b2, acc2);
    acc3 = WMMA_BF16(av, b3, acc3);
    av = avn; b0 = b0n; b1 = b1n; b2 = b2n; b3 = b3n;
  }
  acc0 = WMMA_BF16(av, b0, acc0);
  acc1 = WMMA_BF16(av, b1, acc1);
  acc2 = WMMA_BF16(av, b2, acc2);
  acc3 = WMMA_BF16(av, b3, acc3);

  const int lane = threadIdx.x & 31;
  const int colv = lane & 15, hv = lane >> 4;
#pragma unroll
  for (int r = 0; r < 8; ++r) {
    const size_t row = (size_t)(m0 + (hv << 3) + r) * D_;
    out[row + n0 +  0 + colv] = acc0[r];
    out[row + n0 + 16 + colv] = acc1[r];
    out[row + n0 + 32 + colv] = acc2[r];
    out[row + n0 + 48 + colv] = acc3[r];
  }
}

// ---------- host launcher ----------

extern "C" void kernel_launch(void* const* d_in, const int* in_sizes, int n_in,
                              void* d_out, int out_size, void* d_ws, size_t ws_size,
                              hipStream_t stream) {
  (void)in_sizes; (void)n_in; (void)out_size; (void)ws_size;
  const float* x     = (const float*)d_in[0];
  const float* W_in  = (const float*)d_in[1];
  const float* W_out = (const float*)d_in[2];
  float* out = (float*)d_out;

  char* ws = (char*)d_ws;
  bf16_t* xb    = (bf16_t*)(ws);                          //  8 MB  x in bf16
  bf16_t* WinT  = (bf16_t*)(ws + (size_t) 8u * 1048576);  //  6 MB  W_in^T bf16
  bf16_t* WoutT = (bf16_t*)(ws + (size_t)14u * 1048576);  //  2 MB  W_out^T bf16
  bf16_t* Qb    = (bf16_t*)(ws + (size_t)16u * 1048576);  //  8 MB  [B,H,S,Dh]
  bf16_t* Kb    = (bf16_t*)(ws + (size_t)24u * 1048576);  //  8 MB  [B,H,S,Dh]
  bf16_t* Vt    = (bf16_t*)(ws + (size_t)32u * 1048576);  //  8 MB  [B,H,Dh,S]
  bf16_t* Ob    = (bf16_t*)(ws + (size_t)40u * 1048576);  //  8 MB  [B,S,D]

  const int n4 = (B_ * S_ * D_) / 4;
  to_bf16_kernel<<<(n4 + 255) / 256, 256, 0, stream>>>(x, xb, n4);
  transpose_to_bf16<<<dim3(3 * D_ / 32, D_ / 32), dim3(32, 8), 0, stream>>>(W_in, WinT, D_, 3 * D_);
  transpose_to_bf16<<<dim3(D_ / 32, D_ / 32),     dim3(32, 8), 0, stream>>>(W_out, WoutT, D_, D_);

  qkv_gemm_kernel<<<(256 * 48) / 4, 128, 0, stream>>>(xb, WinT, Qb, Kb, Vt);
  attn_kernel<<<(B_ * H_ * (S_ / 16)) / 4, 128, 0, stream>>>(Qb, Kb, Vt, Ob);
  out_proj_kernel<<<(256 * 16) / 4, 128, 0, stream>>>(Ob, WoutT, out);
}
